// HHDFlow_55585466744908
// MI455X (gfx1250) — compile-verified, matrix operand order checked by reading, not run on hardware
//
#include <hip/hip_runtime.h>
#include <hip/hip_bf16.h>

typedef __attribute__((ext_vector_type(16))) _Float16 v16h;
typedef __attribute__((ext_vector_type(8)))  _Float16 v8h;
typedef __attribute__((ext_vector_type(8)))  float    v8f;
typedef __attribute__((ext_vector_type(4)))  unsigned int v4u;

static inline int cdiv(int a, int b) { return (a + b - 1) / b; }

// ---------------------------------------------------------------------------
// Hot path: 3x3 conv 128->128, stride 1, pad 1, via v_wmma_f32_16x16x32_f16,
// with per-tap weight slabs staged in LDS (double-buffered) so every wmma
// operand is a ds_load (no global-load wait on the critical path).
// Block = 256 thr = 8 waves = {4 rows} x {2 co-halves}; block out = 4x16 px x 128 co.
// in : HWC f16 [S][S][128] ; wh : f16 [128][9][128] ; out: HWC f16 [S][S][128]
// out = leaky(conv + bias) (+ res, post-leaky, if res != null)
// grid = (S/4, S/16); dyn LDS = (6*18*128 + 2*128*128)*2 = 93184 B
// ---------------------------------------------------------------------------
__global__ __launch_bounds__(256)
void conv3x3_wmma_ws(const _Float16* __restrict__ in, const _Float16* __restrict__ wh,
                     const float* __restrict__ bias, _Float16* __restrict__ out,
                     const _Float16* __restrict__ res, int S)
{
    constexpr int CIN  = 128;
    constexpr int CIN8 = CIN / 8;                 // 16B vectors per pixel
    extern __shared__ _Float16 dynsm[];
    _Float16* halo  = dynsm;                      // [6][18][128]
    _Float16* wbuf0 = dynsm + 6 * 18 * CIN;       // [128][128] tap slab (even taps)
    _Float16* wbuf1 = wbuf0 + 128 * CIN;          // [128][128] tap slab (odd taps)

    const int tid = threadIdx.x;
    const int by0 = blockIdx.x << 2;              // first output row of block
    const int x0  = blockIdx.y << 4;              // first output col of tile

    // ---- stage input halo (6 rows x 18 cols) + tap-0 weight slab ----
    for (int i = tid; i < 6 * 18 * CIN8; i += 256) {
        int c8  = i % CIN8;
        int rc  = i / CIN8;
        int col = rc % 18;
        int r   = rc / 18;
        int gy  = by0 + r - 1;
        int gx  = x0 + col - 1;
        v4u v = {0u, 0u, 0u, 0u};
        if (gy >= 0 && gy < S && gx >= 0 && gx < S)
            v = *(const v4u*)(in + ((size_t)gy * S + gx) * CIN + c8 * 8);
        *(v4u*)(halo + (size_t)(r * 18 + col) * CIN + c8 * 8) = v;
    }
#pragma unroll
    for (int k = 0; k < 8; ++k) {                 // 128*16 vectors / 256 thr = 8 each
        int i  = tid + 256 * k;
        int c8 = i % CIN8, co = i / CIN8;
        v4u v = *(const v4u*)(wh + ((size_t)co * 9 + 0) * CIN + c8 * 8);
        *(v4u*)(wbuf0 + (size_t)co * CIN + c8 * 8) = v;
    }
    __syncthreads();

    const int lane = tid & 31;
    const int wave = tid >> 5;
    const int mt   = wave & 3;         // row within block
    const int cg   = wave >> 2;        // 0/1 -> co base 0/64
    const int n    = lane & 15;        // A: M row; B/D: N col
    const int hs   = lane >> 4;        // K half-select (ISA 16-bit A/B layout)
    const int off  = hs << 3;          // 0 or 8 halfs

    v8f acc0 = {}, acc1 = {}, acc2 = {}, acc3 = {};
    const size_t coB = (size_t)(cg * 64 + n) * CIN + off;  // t=0 B base in slab

    for (int tap = 0; tap < 9; ++tap) {
        const int ky = tap / 3, kx = tap % 3;
        const _Float16* cw = (tap & 1) ? wbuf1 : wbuf0;
        _Float16*       nw = (tap & 1) ? wbuf0 : wbuf1;

        // prefetch next tap's weight slab into registers (hidden under the wmmas)
        v4u pf[8];
        if (tap < 8) {
#pragma unroll
            for (int k = 0; k < 8; ++k) {
                int i  = tid + 256 * k;
                int c8 = i % CIN8, co = i / CIN8;
                pf[k] = *(const v4u*)(wh + ((size_t)co * 9 + tap + 1) * CIN + c8 * 8);
            }
        }

        // lanes 0-15: K = ch+0..7 & ch+16..23 ; lanes 16-31: K = ch+8..15 & ch+24..31
        const _Float16* arow = halo + (size_t)((mt + ky) * 18 + (n + kx)) * CIN + off;
#pragma unroll
        for (int ch = 0; ch < CIN; ch += 32) {
            v8h alo = *(const v8h*)(arow + ch);
            v8h ahi = *(const v8h*)(arow + ch + 16);
            v16h A;
#pragma unroll
            for (int i = 0; i < 8; ++i) { A[i] = alo[i]; A[i + 8] = ahi[i]; }
#pragma unroll
            for (int t = 0; t < 4; ++t) {
                const _Float16* bp = cw + coB + (size_t)t * 16 * CIN + ch;
                v8h blo = *(const v8h*)(bp);
                v8h bhi = *(const v8h*)(bp + 16);
                v16h B;
#pragma unroll
                for (int i = 0; i < 8; ++i) { B[i] = blo[i]; B[i + 8] = bhi[i]; }
                v8f& a = (t == 0) ? acc0 : (t == 1) ? acc1 : (t == 2) ? acc2 : acc3;
                a = __builtin_amdgcn_wmma_f32_16x16x32_f16(
                        false, A, false, B, (short)0, a, false, false);
            }
        }

        // flush prefetched slab to the other LDS buffer, then sync for next tap
        if (tap < 8) {
#pragma unroll
            for (int k = 0; k < 8; ++k) {
                int i  = tid + 256 * k;
                int c8 = i % CIN8, co = i / CIN8;
                *(v4u*)(nw + (size_t)co * CIN + c8 * 8) = pf[k];
            }
        }
        __syncthreads();
    }

    // D layout: VGPR g -> M = g + 8*hs, N = lane&15. Bias + leaky (+ residual) fused.
    const int oy = by0 + mt;
#pragma unroll
    for (int t = 0; t < 4; ++t) {
        const int   co = cg * 64 + t * 16 + n;
        const float bv = bias[co];
        const v8f&  a  = (t == 0) ? acc0 : (t == 1) ? acc1 : (t == 2) ? acc2 : acc3;
        _Float16* orow = out + ((size_t)oy * S + x0) * 128 + co;
        const _Float16* rrow = res + ((size_t)oy * S + x0) * 128 + co;
#pragma unroll
        for (int g = 0; g < 8; ++g) {
            int   m = g + (hs << 3);
            float v = a[g] + bv;
            v = (v >= 0.f) ? v : 0.1f * v;
            if (res) v += (float)rrow[(size_t)m * 128];
            orow[(size_t)m * 128] = (_Float16)v;
        }
    }
}

// ---------------------------------------------------------------------------
// First conv (473->128 at 16x16, CIN padded to 480): direct-global B fragments.
// Same tiling as above; this layer is <1% of total FLOPs.
// ---------------------------------------------------------------------------
template <int CIN>
__global__ __launch_bounds__(256)
void conv3x3_wmma(const _Float16* __restrict__ in, const _Float16* __restrict__ wh,
                  const float* __restrict__ bias, _Float16* __restrict__ out,
                  const _Float16* __restrict__ res, int S)
{
    extern __shared__ _Float16 smem[];            // [6][18][CIN] input halo
    const int tid = threadIdx.x;
    const int by0 = blockIdx.x << 2;
    const int x0  = blockIdx.y << 4;

    const int CIN8  = CIN / 8;
    const int total = 6 * 18 * CIN8;
    for (int i = tid; i < total; i += 256) {
        int c8  = i % CIN8;
        int rc  = i / CIN8;
        int col = rc % 18;
        int r   = rc / 18;
        int gy  = by0 + r - 1;
        int gx  = x0 + col - 1;
        v4u v = {0u, 0u, 0u, 0u};
        if (gy >= 0 && gy < S && gx >= 0 && gx < S)
            v = *(const v4u*)(in + ((size_t)gy * S + gx) * CIN + c8 * 8);
        *(v4u*)(smem + (size_t)(r * 18 + col) * CIN + c8 * 8) = v;
    }
    __syncthreads();

    const int lane = tid & 31;
    const int wave = tid >> 5;
    const int mt   = wave & 3;
    const int cg   = wave >> 2;
    const int n    = lane & 15;
    const int hs   = lane >> 4;
    const int off  = hs << 3;

    v8f acc0 = {}, acc1 = {}, acc2 = {}, acc3 = {};
    const _Float16* wbase = wh + (size_t)(cg * 64 + n) * 9 * CIN;

    for (int tap = 0; tap < 9; ++tap) {
        const int ky = tap / 3, kx = tap % 3;
        const _Float16* arow = smem + (size_t)((mt + ky) * 18 + (n + kx)) * CIN;
        const _Float16* brow = wbase + (size_t)tap * CIN;
#pragma unroll
        for (int ch = 0; ch < CIN; ch += 32) {
            v8h alo = *(const v8h*)(arow + ch + off);
            v8h ahi = *(const v8h*)(arow + ch + 16 + off);
            v16h A;
#pragma unroll
            for (int i = 0; i < 8; ++i) { A[i] = alo[i]; A[i + 8] = ahi[i]; }
#pragma unroll
            for (int t = 0; t < 4; ++t) {
                const _Float16* bp = brow + (size_t)t * 16 * 9 * CIN + ch;
                v8h blo = *(const v8h*)(bp + off);
                v8h bhi = *(const v8h*)(bp + 16 + off);
                v16h B;
#pragma unroll
                for (int i = 0; i < 8; ++i) { B[i] = blo[i]; B[i + 8] = bhi[i]; }
                v8f& a = (t == 0) ? acc0 : (t == 1) ? acc1 : (t == 2) ? acc2 : acc3;
                a = __builtin_amdgcn_wmma_f32_16x16x32_f16(
                        false, A, false, B, (short)0, a, false, false);
            }
        }
    }

    const int oy = by0 + mt;
#pragma unroll
    for (int t = 0; t < 4; ++t) {
        const int   co = cg * 64 + t * 16 + n;
        const float bv = bias[co];
        const v8f&  a  = (t == 0) ? acc0 : (t == 1) ? acc1 : (t == 2) ? acc2 : acc3;
        _Float16* orow = out + ((size_t)oy * S + x0) * 128 + co;
        const _Float16* rrow = res + ((size_t)oy * S + x0) * 128 + co;
#pragma unroll
        for (int g = 0; g < 8; ++g) {
            int   m = g + (hs << 3);
            float v = a[g] + bv;
            v = (v >= 0.f) ? v : 0.1f * v;
            if (res) v += (float)rrow[(size_t)m * 128];
            orow[(size_t)m * 128] = (_Float16)v;
        }
    }
}

// ---------------------------------------------------------------------------
// Direct fp32 conv (encoder / redir). in CHW, w OIHW, out CHW (no bias).
// ---------------------------------------------------------------------------
__global__ void conv2d_direct(const float* __restrict__ in, const float* __restrict__ w,
                              float* __restrict__ out,
                              int Cin, int Hin, int Win, int Cout, int Hout, int Wout,
                              int K, int stride, int pad)
{
    int idx = blockIdx.x * blockDim.x + threadIdx.x;
    int tot = Cout * Hout * Wout;
    if (idx >= tot) return;
    int x  = idx % Wout;
    int y  = (idx / Wout) % Hout;
    int co = idx / (Wout * Hout);
    float s = 0.f;
    for (int ci = 0; ci < Cin; ++ci) {
        const float* ip = in + (size_t)ci * Hin * Win;
        const float* wp = w + (((size_t)co * Cin + ci) * K) * K;
        for (int ky = 0; ky < K; ++ky) {
            int yy = y * stride + ky - pad;
            if (yy < 0 || yy >= Hin) continue;
            for (int kx = 0; kx < K; ++kx) {
                int xx = x * stride + kx - pad;
                if (xx < 0 || xx >= Win) continue;
                s += ip[(size_t)yy * Win + xx] * wp[ky * K + kx];
            }
        }
    }
    out[idx] = s;
}

// BatchNorm (training-mode batch stats over H,W; N=1) + LeakyReLU, in place.
__global__ void bn_leaky_kernel(float* __restrict__ x, const float* __restrict__ gamma,
                                const float* __restrict__ beta, int HW)
{
    __shared__ float sh[256], sh2[256];
    int c = blockIdx.x;
    float* p = x + (size_t)c * HW;
    float s = 0.f, s2 = 0.f;
    for (int i = threadIdx.x; i < HW; i += 256) { float v = p[i]; s += v; s2 += v * v; }
    sh[threadIdx.x] = s; sh2[threadIdx.x] = s2;
    __syncthreads();
    for (int o = 128; o > 0; o >>= 1) {
        if (threadIdx.x < o) { sh[threadIdx.x] += sh[threadIdx.x + o];
                               sh2[threadIdx.x] += sh2[threadIdx.x + o]; }
        __syncthreads();
    }
    float m   = sh[0] / HW;
    float var = sh2[0] / HW - m * m;
    float inv = rsqrtf(var + 1e-5f);
    float g = gamma[c], b = beta[c];
    for (int i = threadIdx.x; i < HW; i += 256) {
        float v = (p[i] - m) * inv * g + b;
        p[i] = (v >= 0.f) ? v : 0.1f * v;
    }
}

// FlowNetC cost volume: 21x21 strided displacements, mean over 256 ch, + leaky.
__global__ void correlation_leaky(const float* __restrict__ f1, const float* __restrict__ f2,
                                  float* __restrict__ corr)
{
    int dch = blockIdx.x;
    int pix = threadIdx.x;
    int y = pix >> 4, x = pix & 15;
    int dy = (dch / 21) * 2 - 20;
    int dx = (dch % 21) * 2 - 20;
    int ys = y + dy, xs = x + dx;
    float s = 0.f;
    if (ys >= 0 && ys < 16 && xs >= 0 && xs < 16) {
        int p2 = ys * 16 + xs;
        for (int c = 0; c < 256; ++c) s += f1[c * 256 + pix] * f2[c * 256 + p2];
    }
    s *= (1.f / 256.f);
    s = (s >= 0.f) ? s : 0.1f * s;
    corr[dch * 256 + pix] = s;
}

// feat [16][16][480] f16 HWC: c<32 -> redir, 32<=c<473 -> corr, else 0 (K pad).
__global__ void build_feat(const float* __restrict__ redir, const float* __restrict__ corr,
                           _Float16* __restrict__ feat)
{
    int idx = blockIdx.x * blockDim.x + threadIdx.x;
    if (idx >= 16 * 16 * 480) return;
    int c = idx % 480, pix = idx / 480;
    float v = (c < 32) ? redir[c * 256 + pix]
            : (c < 473) ? corr[(c - 32) * 256 + pix] : 0.f;
    feat[idx] = (_Float16)v;
}

// OIHW fp32 -> [co][tap][c] f16 (c padded to CinPad with zeros).
__global__ void wtrans(const float* __restrict__ w, _Float16* __restrict__ wh,
                       int Cin, int CinPad, int Co)
{
    int idx = blockIdx.x * blockDim.x + threadIdx.x;
    int tot = Co * 9 * CinPad;
    if (idx >= tot) return;
    int c   = idx % CinPad;
    int rem = idx / CinPad;
    int tap = rem % 9;
    int co  = rem / 9;
    wh[idx] = (c < Cin) ? (_Float16)w[((size_t)co * Cin + c) * 9 + tap] : (_Float16)0.f;
}

__global__ void copy_h(const _Float16* __restrict__ a, _Float16* __restrict__ b, int n)
{
    int i = blockIdx.x * blockDim.x + threadIdx.x;
    if (i < n) b[i] = a[i];
}

// nearest 2x upsample, HWC f16, 128 channels; writes x and new x0 in one pass.
__global__ void upsample2x_dual(const _Float16* __restrict__ in, _Float16* __restrict__ o1,
                                _Float16* __restrict__ o2, int S)
{
    int idx = blockIdx.x * blockDim.x + threadIdx.x;
    int n = 4 * S * S * 128;
    if (idx >= n) return;
    int c = idx & 127;
    int p = idx >> 7;
    int X = p % (2 * S);
    int Y = p / (2 * S);
    _Float16 v = in[(((size_t)(Y >> 1)) * S + (X >> 1)) * 128 + c];
    o1[idx] = v;
    o2[idx] = v;
}

// last conv 128->2, 3x3 pad1 at 128x128 + bias + leaky. in HWC f16, out CHW fp32.
__global__ void conv_last(const _Float16* __restrict__ in, const float* __restrict__ w,
                          const float* __restrict__ b, float* __restrict__ out)
{
    int idx = blockIdx.x * blockDim.x + threadIdx.x;
    if (idx >= 2 * 128 * 128) return;
    int x  = idx & 127;
    int y  = (idx >> 7) & 127;
    int cc = idx >> 14;
    float s = b[cc];
    for (int ky = 0; ky < 3; ++ky) {
        int yy = y + ky - 1;
        if (yy < 0 || yy > 127) continue;
        for (int kx = 0; kx < 3; ++kx) {
            int xx = x + kx - 1;
            if (xx < 0 || xx > 127) continue;
            const _Float16* ip = in + ((size_t)yy * 128 + xx) * 128;
            const float*    wp = w + (size_t)cc * 128 * 9 + ky * 3 + kx;
            for (int c = 0; c < 128; ++c) s += (float)ip[c] * wp[(size_t)c * 9];
        }
    }
    s = (s >= 0.f) ? s : 0.1f * s;
    out[idx] = s;
}

// tuple (r+h+d, r, d, h) packed flat.
__global__ void combine(const float* __restrict__ r, const float* __restrict__ h,
                        const float* __restrict__ d, float* __restrict__ out, int n)
{
    int i = blockIdx.x * blockDim.x + threadIdx.x;
    if (i >= n) return;
    float rv = r[i], hv = h[i], dv = d[i];
    out[i]         = rv + hv + dv;
    out[n + i]     = rv;
    out[2 * n + i] = dv;
    out[3 * n + i] = hv;
}

// ---------------------------------------------------------------------------
extern "C" void kernel_launch(void* const* d_in, const int* in_sizes, int n_in,
                              void* d_out, int out_size, void* d_ws, size_t ws_size,
                              hipStream_t stream)
{
    (void)in_sizes; (void)n_in; (void)out_size; (void)ws_size;
    // setup_inputs() insertion order: x1, x2, then params (conv1..bnr), then gen1/2/3.
    const float* x1      = (const float*)d_in[0];
    const float* x2      = (const float*)d_in[1];
    const float* conv1_w = (const float*)d_in[2];
    const float* bn1_g   = (const float*)d_in[3];
    const float* bn1_b   = (const float*)d_in[4];
    const float* conv2_w = (const float*)d_in[5];
    const float* bn2_g   = (const float*)d_in[6];
    const float* bn2_b   = (const float*)d_in[7];
    const float* conv3_w = (const float*)d_in[8];
    const float* bn3_g   = (const float*)d_in[9];
    const float* bn3_b   = (const float*)d_in[10];
    const float* redir_w = (const float*)d_in[11];
    const float* bnr_g   = (const float*)d_in[12];
    const float* bnr_b   = (const float*)d_in[13];

    // ---- workspace carve-up (~22 MB) ----
    char* wsb = (char*)d_ws;
    size_t off = 0;
    auto carve = [&](size_t bytes) -> char* {
        char* p = wsb + off; off += (bytes + 255) & ~(size_t)255; return p;
    };
    float*    c1    = (float*)carve(64ull * 64 * 64 * 4);
    float*    c2    = (float*)carve(128ull * 32 * 32 * 4);
    float*    c3a   = (float*)carve(256ull * 16 * 16 * 4);
    float*    c3b   = (float*)carve(256ull * 16 * 16 * 4);
    float*    redir = (float*)carve(32ull * 16 * 16 * 4);
    float*    corr  = (float*)carve(441ull * 16 * 16 * 4);
    _Float16* feat  = (_Float16*)carve(16ull * 16 * 480 * 2);
    _Float16* whF   = (_Float16*)carve(128ull * 9 * 480 * 2);
    _Float16* whB   = (_Float16*)carve(16ull * 128 * 9 * 128 * 2);
    _Float16* bufA  = (_Float16*)carve(128ull * 128 * 128 * 2);
    _Float16* bufB  = (_Float16*)carve(128ull * 128 * 128 * 2);
    _Float16* bufX0 = (_Float16*)carve(128ull * 128 * 128 * 2);
    float* genOut[3] = { (float*)carve(2ull * 128 * 128 * 4),
                         (float*)carve(2ull * 128 * 128 * 4),
                         (float*)carve(2ull * 128 * 128 * 4) };

    // ---- siamese encoder (shared weights, per-tensor batch-stat BN) ----
    auto enc = [&](const float* img, float* c3) {
        conv2d_direct<<<cdiv(64 * 64 * 64, 256), 256, 0, stream>>>(
            img, conv1_w, c1, 3, 128, 128, 64, 64, 64, 7, 2, 3);
        bn_leaky_kernel<<<64, 256, 0, stream>>>(c1, bn1_g, bn1_b, 64 * 64);
        conv2d_direct<<<cdiv(128 * 32 * 32, 256), 256, 0, stream>>>(
            c1, conv2_w, c2, 64, 64, 64, 128, 32, 32, 5, 2, 2);
        bn_leaky_kernel<<<128, 256, 0, stream>>>(c2, bn2_g, bn2_b, 32 * 32);
        conv2d_direct<<<cdiv(256 * 16 * 16, 256), 256, 0, stream>>>(
            c2, conv3_w, c3, 128, 32, 32, 256, 16, 16, 5, 2, 2);
        bn_leaky_kernel<<<256, 256, 0, stream>>>(c3, bn3_g, bn3_b, 16 * 16);
    };
    enc(x1, c3a);
    enc(x2, c3b);

    conv2d_direct<<<cdiv(32 * 16 * 16, 256), 256, 0, stream>>>(
        c3a, redir_w, redir, 256, 16, 16, 32, 16, 16, 1, 1, 0);
    bn_leaky_kernel<<<32, 256, 0, stream>>>(redir, bnr_g, bnr_b, 256);
    correlation_leaky<<<441, 256, 0, stream>>>(c3a, c3b, corr);
    build_feat<<<cdiv(16 * 16 * 480, 256), 256, 0, stream>>>(redir, corr, feat);

    // ---- three generators (r = gen1, h = gen2, d = gen3) ----
    const size_t WB      = 128ull * 9 * 128;          // per-conv weight size (halfs)
    const size_t LDS_WS  = (6 * 18 * 128 + 2 * 128 * 128) * 2;  // 93184 B
    for (int g = 0; g < 3; ++g) {
        int bi = 14 + g * 36;
        const float* fw = (const float*)d_in[bi + 0];
        const float* fb = (const float*)d_in[bi + 1];
        const float* lw = (const float*)d_in[bi + 2];
        const float* lb = (const float*)d_in[bi + 3];

        wtrans<<<cdiv(128 * 9 * 480, 256), 256, 0, stream>>>(fw, whF, 473, 480, 128);
        for (int j = 0; j < 16; ++j) {
            const float* bw = (const float*)d_in[bi + 4 + 2 * j];
            wtrans<<<cdiv(128 * 9 * 128, 256), 256, 0, stream>>>(
                bw, whB + WB * j, 128, 128, 128);
        }

        // first conv 473(->480) -> 128 at 16x16 (direct-global B path)
        conv3x3_wmma<480><<<dim3(4, 1), 256, 6 * 18 * 480 * 2, stream>>>(
            feat, whF, fb, bufA, (const _Float16*)nullptr, 16);
        copy_h<<<cdiv(16 * 16 * 128, 256), 256, 0, stream>>>(bufA, bufX0, 16 * 16 * 128);

        _Float16* cur = bufA;
        _Float16* alt = bufB;
        int S = 16;
        for (int idx = 0; idx < 4; ++idx) {
            const dim3 grid(S / 4, S / 16);
            const float* b0 = (const float*)d_in[bi + 5 + 2 * (idx * 4 + 0)];
            const float* b1 = (const float*)d_in[bi + 5 + 2 * (idx * 4 + 1)];
            const float* b2 = (const float*)d_in[bi + 5 + 2 * (idx * 4 + 2)];
            const float* b3 = (const float*)d_in[bi + 5 + 2 * (idx * 4 + 3)];
            conv3x3_wmma_ws<<<grid, 256, LDS_WS, stream>>>(
                cur, whB + WB * (idx * 4 + 0), b0, alt, (const _Float16*)nullptr, S);
            conv3x3_wmma_ws<<<grid, 256, LDS_WS, stream>>>(
                alt, whB + WB * (idx * 4 + 1), b1, cur, (const _Float16*)nullptr, S);
            conv3x3_wmma_ws<<<grid, 256, LDS_WS, stream>>>(
                cur, whB + WB * (idx * 4 + 2), b2, alt, (const _Float16*)nullptr, S);
            // 4th conv: fused post-leaky residual (+ x0)
            conv3x3_wmma_ws<<<grid, 256, LDS_WS, stream>>>(
                alt, whB + WB * (idx * 4 + 3), b3, cur, bufX0, S);
            if (idx < 3) {
                upsample2x_dual<<<cdiv(4 * S * S * 128, 256), 256, 0, stream>>>(
                    cur, alt, bufX0, S);
                S *= 2;
                _Float16* t = cur; cur = alt; alt = t;   // x now lives in old 'alt'
            }
        }
        conv_last<<<cdiv(2 * 128 * 128, 256), 256, 0, stream>>>(cur, lw, lb, genOut[g]);
    }

    combine<<<cdiv(2 * 128 * 128, 256), 256, 0, stream>>>(
        genOut[0], genOut[1], genOut[2], (float*)d_out, 2 * 128 * 128);
}
